// BondTypeAwareAttention_24283745091780
// MI455X (gfx1250) — matrix operand order repaired and checked by reading, not compiled, not run on hardware
//
#include <hip/hip_runtime.h>
#include <math.h>

#define B_  32
#define N_  256
#define D_  256
#define H_  8
#define E_  5
#define HD_ 32
#define SCALE_ 0.17677669529663687f   // HD^-0.5

typedef __attribute__((ext_vector_type(16))) _Float16 v16h;
typedef __attribute__((ext_vector_type(8)))  float    v8f;

static __device__ __forceinline__ v8f wmma_f16(v16h a, v16h b, v8f c) {
  // D = A(16x32 f16) * B(32x16 f16) + C(16x16 f32)
  return __builtin_amdgcn_wmma_f32_16x16x32_f16(
      /*neg_a=*/false, a, /*neg_b=*/false, b,
      /*c_mod=*/(short)0, c, /*reuse_a=*/false, /*reuse_b=*/false);
}

// ---- WMMA operand gathers (contiguous k order, k = 16*half + s) ----------------------
// A tile 16(M) x 32(K), source row-major [M][K], leading dim ld (elements).
// lanes 0-15: M=lane, K=0..15 in V0..7 (2/VGPR); lanes 16-31: K=16..31.
template <typename T>
static __device__ __forceinline__ v16h load_a_rm(const T* p, int ld, int lane) {
  const int m = lane & 15, half = lane >> 4;
  v16h r;
#pragma unroll
  for (int v = 0; v < 8; ++v) {
    const int k = half * 16 + 2 * v;
    r[2 * v]     = (_Float16)p[m * ld + k];
    r[2 * v + 1] = (_Float16)p[m * ld + k + 1];
  }
  return r;
}

// B tile 32(K) x 16(N) where B[k][n] = src[n*ld + k] (src row-major [N][K]).
// lanes 0-15: K=0..15 in V0..7 (2/VGPR); lanes 16-31: K=16..31.
template <typename T>
static __device__ __forceinline__ v16h load_b_nk(const T* p, int ld, int lane) {
  const int n = lane & 15, half = lane >> 4;
  v16h r;
#pragma unroll
  for (int v = 0; v < 8; ++v) {
    const int k = half * 16 + 2 * v;
    r[2 * v]     = (_Float16)p[n * ld + k];
    r[2 * v + 1] = (_Float16)p[n * ld + k + 1];
  }
  return r;
}

// CDNA5 async global->LDS copy (16B per lane), tracked by ASYNCcnt.
static __device__ __forceinline__ void async_copy_b128(unsigned lds_off,
                                                       const void* gaddr) {
  asm volatile("global_load_async_to_lds_b128 %0, %1, off"
               :: "v"(lds_off), "v"(gaddr)
               : "memory");
}
static __device__ __forceinline__ void wait_asynccnt0() {
  asm volatile("s_wait_asynccnt 0x0" ::: "memory");
}

// ---- Kernel 1: QKV GEMM  y(BN,768) = x(BN,256) @ qkv_w^T -------------------
// q,k scattered to [B,H,N,HD]; v scattered TRANSPOSED to [B,H,HD,N] so the
// P@V B-operand is contiguous later.  One wave = 16x64 output (A reused 4x).
__global__ __launch_bounds__(32) void qkv_gemm_kernel(
    const float* __restrict__ x, const float* __restrict__ w,
    _Float16* __restrict__ qh, _Float16* __restrict__ kh,
    _Float16* __restrict__ vt) {
  const int lane = threadIdx.x;
  const int m0 = blockIdx.x * 16;   // row in B*N
  const int c0 = blockIdx.y * 64;   // col in 3*D
  v8f acc[4] = {{}, {}, {}, {}};
#pragma unroll
  for (int kt = 0; kt < D_ / 32; ++kt) {
    v16h a = load_a_rm<float>(x + m0 * D_ + kt * 32, D_, lane);
#pragma unroll
    for (int ct = 0; ct < 4; ++ct) {
      v16h b = load_b_nk<float>(w + (c0 + ct * 16) * D_ + kt * 32, D_, lane);
      acc[ct] = wmma_f16(a, b, acc[ct]);
    }
  }
  const int n = lane & 15, half = lane >> 4;
#pragma unroll
  for (int ct = 0; ct < 4; ++ct) {
    const int c = c0 + ct * 16 + n;
    const int sel = c >> 8;            // 0=q,1=k,2=v
    const int h = (c & 255) >> 5;
    const int d = c & 31;
#pragma unroll
    for (int v = 0; v < 8; ++v) {
      const int gm = m0 + v + half * 8;        // b*N + node
      const int bb = gm >> 8, node = gm & 255;
      const _Float16 val = (_Float16)acc[ct][v];
      if (sel == 0)      qh[((bb * H_ + h) * N_ + node) * HD_ + d] = val;
      else if (sel == 1) kh[((bb * H_ + h) * N_ + node) * HD_ + d] = val;
      else               vt[((bb * H_ + h) * HD_ + d) * N_ + node] = val;
    }
  }
}

// ---- Kernel 2: bond type = argmax over E=5 (first max, like jnp.argmax) ----
__global__ void bond_argmax_kernel(const float* __restrict__ ef,
                                   unsigned char* __restrict__ bt) {
  const int idx = blockIdx.x * blockDim.x + threadIdx.x;   // < B*N*N = 2^21
  const float* p = ef + (long long)idx * E_;
  float best = p[0];
  int bi = 0;
#pragma unroll
  for (int t = 1; t < E_; ++t) {
    float vv = p[t];
    if (vv > best) { best = vv; bi = t; }
  }
  bt[idx] = (unsigned char)bi;
}

// ---- Kernel 3: attention for one (b,h) x 16-row tile -----------------------
__global__ __launch_bounds__(32) void attn_kernel(
    const _Float16* __restrict__ qh, const _Float16* __restrict__ kh,
    const _Float16* __restrict__ vt, const unsigned char* __restrict__ bt,
    const int* __restrict__ node_mask, const float* __restrict__ bond_emb,
    const float* __restrict__ bond_pos_emb, float* __restrict__ interm) {
  __shared__ float         sS[16][N_];   // f32 scores (16 KB)
  __shared__ _Float16      sP[16][N_];   // f16 probabilities (8 KB)
  __shared__ float         sQpos[16][E_];
  __shared__ unsigned char sBT[16 * N_]; // bond-type tile (4 KB, async-filled)

  const int lane = threadIdx.x;
  const int bh = blockIdx.x;             // b*H + h
  const int b = bh >> 3, h = bh & 7;
  const int i0 = blockIdx.y * 16;
  const _Float16* qbase = qh + (bh * N_ + i0) * HD_;

  // --- async prefetch of bond-type rows i0..i0+15 into LDS (4 KB) ----------
  {
    const unsigned char* src = bt + (size_t)(b * N_ + i0) * N_;
    const unsigned lds0 = (unsigned)(size_t)(&sBT[0]);
#pragma unroll
    for (int w = 0; w < 8; ++w) {
      const unsigned off = (unsigned)(lane * 16 + w * 512);
      async_copy_b128(lds0 + off, src + off);
    }
  }

  // --- per-row q . bond_pos_emb[t], split over wave halves -----------------
  {
    const int row = lane & 15, half = lane >> 4;
#pragma unroll
    for (int t = 0; t < E_; ++t) {
      float s = 0.f;
      for (int d = 0; d < HD_ / 2; ++d) {
        const int dd = half * (HD_ / 2) + d;
        s += (float)qbase[row * HD_ + dd] * bond_pos_emb[t * HD_ + dd];
      }
      s += __shfl_xor(s, 16);
      if (half == 0) sQpos[row][t] = s;
    }
  }

  // --- scores(16 x 256) = q @ k^T ; K = HD = 32 ----------------------------
  const v16h aq = load_a_rm<_Float16>(qbase, HD_, lane);
  const int n = lane & 15, half = lane >> 4;
  for (int jt = 0; jt < N_ / 16; ++jt) {
    if (jt + 1 < N_ / 16)
      __builtin_prefetch(kh + (bh * N_ + (jt + 1) * 16) * HD_ + lane * 16, 0, 0);
    v16h bk = load_b_nk<_Float16>(kh + (bh * N_ + jt * 16) * HD_, HD_, lane);
    v8f acc = {};
    acc = wmma_f16(aq, bk, acc);
#pragma unroll
    for (int v = 0; v < 8; ++v)
      sS[v + half * 8][jt * 16 + n] = acc[v];
  }
  wait_asynccnt0();
  __syncthreads();

  // --- row softmax, all 32 lanes: lane owns columns j0..j0+7 ---------------
  {
    const int j0 = lane * 8;
    unsigned mjbits = 0;
#pragma unroll
    for (int c = 0; c < 8; ++c)
      if (node_mask[b * N_ + j0 + c] != 0) mjbits |= (1u << c);
    const bool anyj = __ballot(mjbits != 0u) != 0ull;
    float biasv[E_];
#pragma unroll
    for (int t = 0; t < E_; ++t) biasv[t] = bond_emb[t * H_ + h];

    for (int m = 0; m < 16; ++m) {
      const bool mi = node_mask[b * N_ + i0 + m] != 0;
      float qposv[E_];
#pragma unroll
      for (int t = 0; t < E_; ++t) qposv[t] = sQpos[m][t];
      float s[8];
      float mx = -INFINITY;
#pragma unroll
      for (int c = 0; c < 8; ++c) {
        const int t = sBT[m * N_ + j0 + c];
        float val = (sS[m][j0 + c] + qposv[t]) * SCALE_ + biasv[t];
        val = (mi && ((mjbits >> c) & 1u)) ? val : -INFINITY;
        s[c] = val;
        mx = fmaxf(mx, val);
      }
#pragma unroll
      for (int off = 16; off >= 1; off >>= 1)
        mx = fmaxf(mx, __shfl_xor(mx, off));
      if (mi && anyj) {
        float sum = 0.f;
#pragma unroll
        for (int c = 0; c < 8; ++c) {
          const float e = __expf(s[c] - mx);
          s[c] = e;
          sum += e;
        }
#pragma unroll
        for (int off = 16; off >= 1; off >>= 1)
          sum += __shfl_xor(sum, off);
        const float inv = 1.f / sum;
#pragma unroll
        for (int c = 0; c < 8; ++c)
          sP[m][j0 + c] = (_Float16)(s[c] * inv);
      } else {
        // reference: invalid row -> zero scores -> uniform softmax
#pragma unroll
        for (int c = 0; c < 8; ++c)
          sP[m][j0 + c] = (_Float16)(1.0f / N_);
      }
    }
  }
  __syncthreads();

  // --- out(16 x 32) = P(16 x 256) @ V(256 x 32); V^T layout -> contiguous --
#pragma unroll
  for (int dt = 0; dt < 2; ++dt) {
    v8f acc = {};
#pragma unroll
    for (int kt = 0; kt < N_ / 32; ++kt) {
      v16h ap = load_a_rm<_Float16>(&sP[0][0] + kt * 32, N_, lane);
      v16h bv = load_b_nk<_Float16>(vt + (bh * HD_ + dt * 16) * N_ + kt * 32,
                                    N_, lane);
      acc = wmma_f16(ap, bv, acc);
    }
#pragma unroll
    for (int v = 0; v < 8; ++v) {
      const int irow = i0 + v + half * 8;
      // interm layout (B,N,H,HD) == (B,N,D) after reshape
      interm[((b * N_ + irow) * H_ + h) * HD_ + dt * 16 + n] = acc[v];
    }
  }
}

// ---- Kernel 4: projection  out = interm @ proj_w^T + b ---------------------
__global__ __launch_bounds__(32) void proj_gemm_kernel(
    const float* __restrict__ interm, const float* __restrict__ w,
    const float* __restrict__ bias, float* __restrict__ out) {
  const int lane = threadIdx.x;
  const int m0 = blockIdx.x * 16;
  const int c0 = blockIdx.y * 64;
  v8f acc[4] = {{}, {}, {}, {}};
#pragma unroll
  for (int kt = 0; kt < D_ / 32; ++kt) {
    v16h a = load_a_rm<float>(interm + m0 * D_ + kt * 32, D_, lane);
#pragma unroll
    for (int ct = 0; ct < 4; ++ct) {
      v16h b = load_b_nk<float>(w + (c0 + ct * 16) * D_ + kt * 32, D_, lane);
      acc[ct] = wmma_f16(a, b, acc[ct]);
    }
  }
  const int n = lane & 15, half = lane >> 4;
#pragma unroll
  for (int ct = 0; ct < 4; ++ct) {
    const int c = c0 + ct * 16 + n;
    const float bv = bias[c];
#pragma unroll
    for (int v = 0; v < 8; ++v)
      out[(m0 + v + half * 8) * D_ + c] = acc[ct][v] + bv;
  }
}

extern "C" void kernel_launch(void* const* d_in, const int* in_sizes, int n_in,
                              void* d_out, int out_size, void* d_ws,
                              size_t ws_size, hipStream_t stream) {
  (void)in_sizes; (void)n_in; (void)out_size; (void)ws_size;
  const float* x    = (const float*)d_in[0];   // (B,N,D)
  const float* ef   = (const float*)d_in[1];   // (B,N,N,E)
  const int*   nm   = (const int*)d_in[2];     // (B,N) bool
  const float* qkvw = (const float*)d_in[3];   // (3D,D)
  const float* pw   = (const float*)d_in[4];   // (D,D)
  const float* pb   = (const float*)d_in[5];   // (D,)
  const float* be   = (const float*)d_in[6];   // (E,H)
  const float* bpe  = (const float*)d_in[7];   // (E,HD)
  float* out = (float*)d_out;                  // (B,N,D)

  char* ws = (char*)d_ws;
  size_t off = 0;
  _Float16* qh = (_Float16*)(ws + off); off += (size_t)B_ * H_ * N_ * HD_ * 2;
  _Float16* kh = (_Float16*)(ws + off); off += (size_t)B_ * H_ * N_ * HD_ * 2;
  _Float16* vt = (_Float16*)(ws + off); off += (size_t)B_ * H_ * N_ * HD_ * 2;
  unsigned char* bt = (unsigned char*)(ws + off); off += (size_t)B_ * N_ * N_;
  float* interm = (float*)(ws + off); off += (size_t)B_ * N_ * D_ * 4;

  qkv_gemm_kernel<<<dim3(B_ * N_ / 16, 3 * D_ / 64), 32, 0, stream>>>(
      x, qkvw, qh, kh, vt);
  bond_argmax_kernel<<<(B_ * N_ * N_) / 256, 256, 0, stream>>>(ef, bt);
  attn_kernel<<<dim3(B_ * H_, N_ / 16), 32, 0, stream>>>(
      qh, kh, vt, bt, nm, be, bpe, interm);
  proj_gemm_kernel<<<dim3(B_ * N_ / 16, D_ / 64), 32, 0, stream>>>(
      interm, pw, pb, out);
}
